// TreeMessagePassingLayer_81724637708478
// MI455X (gfx1250) — matrix operand order repaired and checked by reading, not compiled, not run on hardware
//
#include <hip/hip_runtime.h>

// ---------------------------------------------------------------------------
// CDNA5 (gfx1250) WMMA types/helpers
// ---------------------------------------------------------------------------
typedef __attribute__((ext_vector_type(8)))  float  v8f;
typedef __attribute__((ext_vector_type(16))) __bf16 v16bf;
typedef unsigned short ushort_t;
typedef int v4i __attribute__((vector_size(16)));

union Frag { v16bf v; unsigned int u[8]; };

__device__ __forceinline__ ushort_t f2bf(float f) {
  union { float f; unsigned u; } x; x.f = f;
  unsigned r = x.u + 0x7fffu + ((x.u >> 16) & 1u);   // round-to-nearest-even
  return (ushort_t)(r >> 16);
}

__device__ __forceinline__ float gelu_exact(float x) {
  return 0.5f * x * (1.f + erff(x * 0.7071067811865475f));
}

// --- CDNA5 async global->LDS copy (ASYNCcnt-tracked, bypasses VGPRs) -------
__device__ __forceinline__ void async_copy_b128(void* lds_dst, const void* gsrc) {
#if __has_builtin(__builtin_amdgcn_global_load_async_to_lds_b128)
  __builtin_amdgcn_global_load_async_to_lds_b128(
      (__attribute__((address_space(1))) v4i*)gsrc,
      (__attribute__((address_space(3))) v4i*)lds_dst, 0, 0);
#else
  unsigned lo = (unsigned)(unsigned long long)
      (__attribute__((address_space(3))) void*)lds_dst;
  asm volatile("global_load_async_to_lds_b128 %0, %1, off"
               :: "v"(lo), "v"((unsigned long long)gsrc) : "memory");
#endif
}

__device__ __forceinline__ void wait_async0() {
#if __has_builtin(__builtin_amdgcn_s_wait_asynccnt)
  __builtin_amdgcn_s_wait_asynccnt(0);
#else
  asm volatile("s_wait_asynccnt 0" ::: "memory");
#endif
}

// ---------------------------------------------------------------------------
// Generic WMMA GEMM (bf16 in, bf16 or f32 out):
//   out[M,N] = act( A[M,K] @ W[N,K]^T + bias[N] )
// A is a virtual concat of up to 4 bf16 sources, each [M,512] (K = 512*nsrc).
// 128 threads (4 waves, 2x2), block tile 64x64, wave tile 32x32.
// Double-buffered LDS fed by async global->LDS B128 copies.
// ---------------------------------------------------------------------------
#define GSTR 48   // LDS row stride (ushorts): 96B rows, 16B-aligned

__global__ __launch_bounds__(128) void gemm_kernel(
    const ushort_t* __restrict__ A0, const ushort_t* __restrict__ A1,
    const ushort_t* __restrict__ A2, const ushort_t* __restrict__ A3,
    const ushort_t* __restrict__ W,  const float* __restrict__ bias,
    void* __restrict__ outv, int Nout, int K, int act, int out_f32)
{
  __shared__ ushort_t At[2][64 * GSTR];
  __shared__ ushort_t Wt[2][64 * GSTR];

  const ushort_t* srcs[4] = {A0, A1, A2, A3};
  int row0 = blockIdx.y * 64;
  int col0 = blockIdx.x * 64;
  int tid  = threadIdx.x;
  int wave = tid >> 5, lane = tid & 31;
  int wr = wave >> 1, wc = wave & 1;
  int half = lane >> 4, l15 = lane & 15;

  const v8f vzero = {0.f,0.f,0.f,0.f,0.f,0.f,0.f,0.f};
  v8f acc[2][2];
  acc[0][0] = vzero; acc[0][1] = vzero; acc[1][0] = vzero; acc[1][1] = vzero;

  // issue async copies of the 64x32 A and W tiles for K-offset kg into buf
  auto issue = [&](int kg, int buf) {
    const ushort_t* Asrc = srcs[kg >> 9] + (kg & 511);
    for (int e = tid; e < 256; e += 128) {          // 64 rows x 4 16B chunks
      int r = e >> 2, c = e & 3;
      async_copy_b128(&At[buf][r * GSTR + c * 8],
                      Asrc + (size_t)(row0 + r) * 512 + c * 8);
    }
    for (int e = tid; e < 256; e += 128) {
      int r = e >> 2, c = e & 3;
      async_copy_b128(&Wt[buf][r * GSTR + c * 8],
                      W + (size_t)(col0 + r) * K + kg + c * 8);
    }
  };

  issue(0, 0);
  int nk = K >> 5;
  for (int it = 0; it < nk; ++it) {
    wait_async0();
    __syncthreads();
    if (it + 1 < nk) issue((it + 1) << 5, (it + 1) & 1);
    int buf = it & 1;

    Frag a[2], b[2];
    for (int i = 0; i < 2; ++i) {
      int r  = wr * 32 + i * 16 + l15;
      int kb = half * 8;
      for (int v = 0; v < 4; ++v)
        a[i].u[v]     = *(const unsigned int*)&At[buf][r * GSTR + kb + 2 * v];
      for (int v = 0; v < 4; ++v)
        a[i].u[4 + v] = *(const unsigned int*)&At[buf][r * GSTR + kb + 16 + 2 * v];
    }
    for (int j = 0; j < 2; ++j) {
      int n  = wc * 32 + j * 16 + l15;
      int kb = half * 8;
      for (int v = 0; v < 8; ++v)
        b[j].u[v] = *(const unsigned int*)&Wt[buf][n * GSTR + kb + ((v < 4) ? 2 * v : (16 + 2 * (v - 4)))];
    }
    for (int i = 0; i < 2; ++i)
      for (int j = 0; j < 2; ++j)
        acc[i][j] = __builtin_amdgcn_wmma_f32_16x16x32_bf16(
            false, a[i].v, false, b[j].v, (short)0, acc[i][j], false, false);
  }

  for (int i = 0; i < 2; ++i)
    for (int j = 0; j < 2; ++j) {
      int col  = col0 + wc * 32 + j * 16 + l15;
      float bv = bias ? bias[col] : 0.f;
      for (int v = 0; v < 8; ++v) {
        int row = row0 + wr * 32 + i * 16 + v + 8 * half;
        float x = acc[i][j][v] + bv;
        if (act == 1)      x = gelu_exact(x);
        else if (act == 2) x = 1.f / (1.f + __expf(-x));
        size_t idx = (size_t)row * Nout + col;
        if (out_f32) ((float*)outv)[idx] = x;
        else         ((ushort_t*)outv)[idx] = f2bf(x);
      }
    }
}

// ---------------------------------------------------------------------------
// WMMA attention (bf16 Q/K/V in global, bf16 ctx out).
// One block per (b, h, 64-query tile). 128 threads, 4 waves x 16 queries.
// N=128, head_dim=64, NH=8 hardcoded.
// ---------------------------------------------------------------------------
template <bool MASKED>
__global__ __launch_bounds__(128) void attn_kernel(
    const ushort_t* __restrict__ Qp, int ldq, int qoff,
    const ushort_t* __restrict__ Kp, int ldk, int koff,
    const ushort_t* __restrict__ Vp, int ldv, int voff,
    const int* __restrict__ heads, const int* __restrict__ maskp,
    ushort_t* __restrict__ ctx, float* __restrict__ flags)
{
  __shared__ ushort_t Kt[128 * 64];     // [key][d]   (async-loaded)
  __shared__ ushort_t Vt[64 * 128];     // [d][key]   (transposed copy)
  __shared__ ushort_t Pt[4][16 * 132];  // per-wave softmax probs
  __shared__ int sheads[128];
  __shared__ int smask[128];

  int blk  = blockIdx.x;
  int qt   = blk & 1;
  int h    = (blk >> 1) & 7;
  int b    = blk >> 4;
  int tid  = threadIdx.x;
  int wave = tid >> 5, lane = tid & 31;
  int half = lane >> 4, l15 = lane & 15;

  // K tile via async global->LDS (128 rows x 8 16B chunks)
  for (int e = tid; e < 1024; e += 128) {
    int key = e >> 3, c = e & 7;
    async_copy_b128(&Kt[key * 64 + c * 8],
                    Kp + (size_t)(b * 128 + key) * ldk + koff + h * 64 + c * 8);
  }
  // V tile transposed into LDS (per-thread one key row)
  {
    int key = tid;
    const ushort_t* Vrow = Vp + (size_t)(b * 128 + key) * ldv + voff + h * 64;
    for (int c = 0; c < 8; ++c) {
      uint4 a = *reinterpret_cast<const uint4*>(Vrow + c * 8);
      const ushort_t* e16 = (const ushort_t*)&a;
      for (int j = 0; j < 8; ++j)
        Vt[(c * 8 + j) * 128 + key] = e16[j];
    }
  }
  if (MASKED) {
    if (tid < 128) {
      int hh = heads[b * 128 + tid];
      sheads[tid] = hh < 0 ? 0 : (hh > 127 ? 127 : hh);
      smask[tid]  = maskp[b * 128 + tid];
    }
  }
  wait_async0();
  __syncthreads();

  int qbase = qt * 64 + wave * 16;

  // Q fragments (2 K-chunks of 32 over head_dim 64) straight from global bf16
  Frag aq[2];
  {
    int q = qbase + l15;
    const ushort_t* Qrow = Qp + (size_t)(b * 128 + q) * ldq + qoff + h * 64;
    for (int c = 0; c < 2; ++c) {
      int kb = c * 32 + half * 8;
      for (int v = 0; v < 8; ++v) {
        int d = kb + ((v < 4) ? 2 * v : (16 + 2 * (v - 4)));
        aq[c].u[v] = *(const unsigned int*)(Qrow + d);
      }
    }
  }

  const v8f vzero = {0.f,0.f,0.f,0.f,0.f,0.f,0.f,0.f};
  float s[8][8];  // [key col-tile][vgpr]
  for (int ct = 0; ct < 8; ++ct) {
    Frag bk0, bk1;
    int key = ct * 16 + l15;
    for (int v = 0; v < 8; ++v) {
      bk0.u[v] = *(const unsigned int*)&Kt[key * 64 + half * 8 + ((v < 4) ? 2 * v : (16 + 2 * (v - 4)))];
      bk1.u[v] = *(const unsigned int*)&Kt[key * 64 + 32 + half * 8 + ((v < 4) ? 2 * v : (16 + 2 * (v - 4)))];
    }
    v8f acc = vzero;
    acc = __builtin_amdgcn_wmma_f32_16x16x32_bf16(false, aq[0].v, false, bk0.v,
                                                  (short)0, acc, false, false);
    acc = __builtin_amdgcn_wmma_f32_16x16x32_bf16(false, aq[1].v, false, bk1.v,
                                                  (short)0, acc, false, false);
    for (int v = 0; v < 8; ++v) s[ct][v] = acc[v] * 0.125f;  // 1/sqrt(64)
  }

  if (MASKED) {
    for (int ct = 0; ct < 8; ++ct) {
      int k  = ct * 16 + l15;
      int hk = sheads[k], mk = smask[k];
      for (int v = 0; v < 8; ++v) {
        int q = qbase + v + 8 * half;
        bool ok = (sheads[q] == hk) && (q != k) && (smask[q] != 0) && (mk != 0);
        if (!ok) s[ct][v] = -1e30f;
      }
    }
  }

  // row softmax: row r = v + 8*half, cols spread over 16 lanes of the half
  float rinv[8];
  for (int v = 0; v < 8; ++v) {
    float m = s[0][v];
    for (int ct = 1; ct < 8; ++ct) m = fmaxf(m, s[ct][v]);
    for (int off = 1; off < 16; off <<= 1) m = fmaxf(m, __shfl_xor(m, off, 32));
    bool rowvalid = !MASKED || (m > -1e29f);
    float sum = 0.f;
    for (int ct = 0; ct < 8; ++ct) {
      float p = rowvalid ? __expf(s[ct][v] - m) : 0.f;
      s[ct][v] = p;
      sum += p;
    }
    for (int off = 1; off < 16; off <<= 1) sum += __shfl_xor(sum, off, 32);
    rinv[v] = (sum > 0.f) ? (1.f / sum) : 0.f;
    if (MASKED) {
      if (l15 == 0)
        flags[(size_t)b * 128 + qbase + v + 8 * half] = rowvalid ? 1.f : 0.f;
    }
  }

  // stage P (bf16) in per-wave LDS, re-read as A fragments (same-wave DS order)
  for (int ct = 0; ct < 8; ++ct)
    for (int v = 0; v < 8; ++v)
      Pt[wave][(v + 8 * half) * 132 + ct * 16 + l15] = f2bf(s[ct][v] * rinv[v]);

  for (int dt = 0; dt < 4; ++dt) {
    v8f o = vzero;
    for (int kc = 0; kc < 4; ++kc) {
      Frag ap, bv;
      int kbA = kc * 32 + half * 8;
      for (int v = 0; v < 8; ++v) {
        int k = kbA + ((v < 4) ? 2 * v : (16 + 2 * (v - 4)));
        ap.u[v] = *(const unsigned int*)&Pt[wave][l15 * 132 + k];
      }
      int d   = dt * 16 + l15;
      int kbB = kc * 32 + half * 8;
      for (int v = 0; v < 8; ++v)
        bv.u[v] = *(const unsigned int*)&Vt[d * 128 + kbB + ((v < 4) ? 2 * v : (16 + 2 * (v - 4)))];
      o = __builtin_amdgcn_wmma_f32_16x16x32_bf16(false, ap.v, false, bv.v,
                                                  (short)0, o, false, false);
    }
    for (int v = 0; v < 8; ++v) {
      int q = qbase + v + 8 * half;
      ctx[(size_t)(b * 128 + q) * 512 + h * 64 + dt * 16 + l15] = f2bf(o[v]);
    }
  }
}

// ---------------------------------------------------------------------------
// Elementwise helper kernels
// ---------------------------------------------------------------------------
__global__ void cvt_bf16_kernel(const float* __restrict__ src,
                                ushort_t* __restrict__ dst, int n) {
  int i = blockIdx.x * blockDim.x + threadIdx.x;
  if (i < n) dst[i] = f2bf(src[i]);
}

__global__ void gather_kernel(const ushort_t* __restrict__ Hb,
                              const int* __restrict__ pred,
                              ushort_t* __restrict__ HR, int total) {
  int i = blockIdx.x * blockDim.x + threadIdx.x;
  if (i >= total) return;
  int d = i & 511, bn = i >> 9, b = bn >> 7;
  int h = pred[bn]; h = h < 0 ? 0 : (h > 127 ? 127 : h);
  HR[i] = Hb[((size_t)(b * 128 + h) << 9) + d];
}

__global__ void zero_kernel(float* __restrict__ p, int n) {
  int i = blockIdx.x * blockDim.x + threadIdx.x;
  if (i < n) p[i] = 0.f;
}

__global__ void scatter_kernel(const float* __restrict__ H,
                               const int* __restrict__ pred,
                               const int* __restrict__ maskp,
                               float* __restrict__ csum,
                               float* __restrict__ ccnt) {
  int bn = blockIdx.x;
  int b = bn >> 7, n = bn & 127;
  if (maskp[bn] == 0 || n < 1) return;
  int t = pred[bn]; t = t < 0 ? 0 : (t > 127 ? 127 : t);
  size_t src = (size_t)bn * 512, dst = (size_t)(b * 128 + t) * 512;
  for (int d = threadIdx.x; d < 512; d += blockDim.x)
    atomicAdd(&csum[dst + d], H[src + d]);
  if (threadIdx.x == 0) atomicAdd(&ccnt[b * 128 + t], 1.f);
}

__global__ void avg_kernel(const float* __restrict__ csum,
                           const float* __restrict__ ccnt,
                           ushort_t* __restrict__ cavg, int total) {
  int i = blockIdx.x * blockDim.x + threadIdx.x;
  if (i < total) cavg[i] = f2bf(csum[i] / fmaxf(ccnt[i >> 9], 1.f));
}

__global__ void rowzero_kernel(ushort_t* __restrict__ sm,
                               const float* __restrict__ flags, int total) {
  int i = blockIdx.x * blockDim.x + threadIdx.x;
  if (i < total && flags[i >> 9] == 0.f) sm[i] = 0;
}

// out = layernorm(H + gate*update);  one block per row of 512
__global__ __launch_bounds__(256) void fusion_ln_kernel(
    const float* __restrict__ H, const float* __restrict__ G,
    const float* __restrict__ U, const float* __restrict__ lg,
    const float* __restrict__ lb, float* __restrict__ out) {
  size_t base = (size_t)blockIdx.x * 512;
  int tid = threadIdx.x;
  float x0 = H[base + tid]       + G[base + tid]       * U[base + tid];
  float x1 = H[base + tid + 256] + G[base + tid + 256] * U[base + tid + 256];
  float s = x0 + x1, s2 = x0 * x0 + x1 * x1;
  __shared__ float ws1[8], ws2[8];
  for (int m = 16; m >= 1; m >>= 1) {
    s  += __shfl_xor(s, m, 32);
    s2 += __shfl_xor(s2, m, 32);
  }
  int wv = tid >> 5, ln = tid & 31;
  if (ln == 0) { ws1[wv] = s; ws2[wv] = s2; }
  __syncthreads();
  if (tid == 0) {
    float a = 0.f, c = 0.f;
    for (int i = 0; i < 8; ++i) { a += ws1[i]; c += ws2[i]; }
    ws1[0] = a; ws2[0] = c;
  }
  __syncthreads();
  float mu  = ws1[0] * (1.f / 512.f);
  float var = ws2[0] * (1.f / 512.f) - mu * mu;
  float inv = rsqrtf(var + 1e-5f);
  out[base + tid]       = (x0 - mu) * inv * lg[tid]       + lb[tid];
  out[base + tid + 256] = (x1 - mu) * inv * lg[tid + 256] + lb[tid + 256];
}

// ---------------------------------------------------------------------------
// Orchestration
// ---------------------------------------------------------------------------
extern "C" void kernel_launch(void* const* d_in, const int* in_sizes, int n_in,
                              void* d_out, int out_size, void* d_ws, size_t ws_size,
                              hipStream_t stream) {
  const float* H          = (const float*)d_in[0];
  const int*   pred       = (const int*)  d_in[1];
  const int*   maskp      = (const int*)  d_in[2];
  const float* head_in_w  = (const float*)d_in[3];
  const float* head_in_b  = (const float*)d_in[4];
  const float* head_out_w = (const float*)d_in[5];
  const float* head_out_b = (const float*)d_in[6];
  const float* sib_in_w   = (const float*)d_in[7];
  const float* sib_in_b   = (const float*)d_in[8];
  const float* sib_out_w  = (const float*)d_in[9];
  const float* sib_out_b  = (const float*)d_in[10];
  const float* child_w    = (const float*)d_in[11];
  const float* child_b    = (const float*)d_in[12];
  const float* gate_w     = (const float*)d_in[13];
  const float* gate_b     = (const float*)d_in[14];
  const float* trans_w    = (const float*)d_in[15];
  const float* trans_b    = (const float*)d_in[16];
  const float* ln_g       = (const float*)d_in[17];
  const float* ln_b       = (const float*)d_in[18];
  float* out = (float*)d_out;

  const size_t NT = 16384;  // B*N tokens
  char* p = (char*)d_ws;
  auto aus = [&](size_t n) { ushort_t* r = (ushort_t*)p; p += n * 2; return r; };
  auto af  = [&](size_t n) { float*    r = (float*)p;    p += n * 4; return r; };

  // bf16 activations
  ushort_t* Hb    = aus(NT * 512);
  ushort_t* HRb   = aus(NT * 512);
  ushort_t* QHb   = aus(NT * 512);
  ushort_t* KVHb  = aus(NT * 1024);
  ushort_t* QKVSb = aus(NT * 1536);
  ushort_t* CTXHb = aus(NT * 512);
  ushort_t* CTXSb = aus(NT * 512);
  ushort_t* HMb   = aus(NT * 512);
  ushort_t* SMb   = aus(NT * 512);
  ushort_t* CMb   = aus(NT * 512);
  ushort_t* CAVGb = aus(NT * 512);
  // bf16 weights
  ushort_t* hiwb = aus(1536 * 512);
  ushort_t* howb = aus(512 * 512);
  ushort_t* siwb = aus(1536 * 512);
  ushort_t* sowb = aus(512 * 512);
  ushort_t* cwb  = aus(512 * 1024);
  ushort_t* gwb  = aus(512 * 2048);
  ushort_t* twb  = aus(512 * 2048);
  // f32 scratch
  float* CSUM  = af(NT * 512);
  float* CCNT  = af(NT);
  float* FLAGS = af(NT);
  float* G     = af(NT * 512);
  float* U     = af(NT * 512);

  auto cvt = [&](const float* s, ushort_t* d, int n) {
    cvt_bf16_kernel<<<(n + 255) / 256, 256, 0, stream>>>(s, d, n);
  };
  auto gemm = [&](const ushort_t* A0, const ushort_t* A1, const ushort_t* A2,
                  const ushort_t* A3, const ushort_t* W, const float* bias,
                  void* O, int Nout, int K, int act, int out_f32) {
    dim3 grid(Nout / 64, 16384 / 64);
    gemm_kernel<<<grid, 128, 0, stream>>>(A0, A1, A2, A3, W, bias, O,
                                          Nout, K, act, out_f32);
  };

  // 0) one-time f32 -> bf16 conversions (H + all GEMM weights)
  cvt(H, Hb, (int)(NT * 512));
  cvt(head_in_w,  hiwb, 1536 * 512);
  cvt(head_out_w, howb, 512 * 512);
  cvt(sib_in_w,   siwb, 1536 * 512);
  cvt(sib_out_w,  sowb, 512 * 512);
  cvt(child_w,    cwb,  512 * 1024);
  cvt(gate_w,     gwb,  512 * 2048);
  cvt(trans_w,    twb,  512 * 2048);

  // 1) head_repr gather + projections
  gather_kernel<<<32768, 256, 0, stream>>>(Hb, pred, HRb, (int)(NT * 512));
  gemm(Hb,  0, 0, 0, hiwb,              head_in_b,       QHb,   512,  512, 0, 0);
  gemm(HRb, 0, 0, 0, hiwb + 512 * 512,  head_in_b + 512, KVHb,  1024, 512, 0, 0);
  gemm(Hb,  0, 0, 0, siwb,              sib_in_b,        QKVSb, 1536, 512, 0, 0);

  // 2) attention cores (B*NH*2 blocks)
  attn_kernel<false><<<2048, 128, 0, stream>>>(QHb, 512, 0, KVHb, 1024, 0,
                                               KVHb, 1024, 512, nullptr, nullptr,
                                               CTXHb, nullptr);
  gemm(CTXHb, 0, 0, 0, howb, head_out_b, HMb, 512, 512, 0, 0);

  attn_kernel<true><<<2048, 128, 0, stream>>>(QKVSb, 1536, 0, QKVSb, 1536, 512,
                                              QKVSb, 1536, 1024, pred, maskp,
                                              CTXSb, FLAGS);
  gemm(CTXSb, 0, 0, 0, sowb, sib_out_b, SMb, 512, 512, 0, 0);
  rowzero_kernel<<<32768, 256, 0, stream>>>(SMb, FLAGS, (int)(NT * 512));

  // 3) child aggregation (f32 atomics, then averaged into bf16)
  zero_kernel<<<(int)((NT * 512 + NT + 255) / 256), 256, 0, stream>>>(
      CSUM, (int)(NT * 512 + NT));  // CSUM and CCNT are contiguous
  scatter_kernel<<<16384, 128, 0, stream>>>(H, pred, maskp, CSUM, CCNT);
  avg_kernel<<<32768, 256, 0, stream>>>(CSUM, CCNT, CAVGb, (int)(NT * 512));
  gemm(Hb, CAVGb, 0, 0, cwb, child_b, CMb, 512, 1024, 1, 0);  // gelu

  // 4) gated fusion: combined = [H, head_msg, child_msg, sib_msg] (virtual)
  gemm(Hb, HMb, CMb, SMb, gwb, gate_b,  G, 512, 2048, 2, 1);  // sigmoid, f32
  gemm(Hb, HMb, CMb, SMb, twb, trans_b, U, 512, 2048, 1, 1);  // gelu, f32

  // 5) residual + layernorm (f32)
  fusion_ln_kernel<<<16384, 256, 0, stream>>>(H, G, U, ln_g, ln_b, out);
}
